// DeformableConv2d_29721173688772
// MI455X (gfx1250) — compile-verified, matrix operand order checked by reading, not run on hardware
//
#include <hip/hip_runtime.h>

// Problem constants (from reference setup_inputs)
#define Bv   4
#define Cv   64
#define Hv   160
#define Wv   160
#define Ov   64
#define Gv   4
#define CGv  16
#define KKv  9
#define HWv  (Hv * Wv)          // 25600
#define Kv   (Gv * CGv * KKv)   // 576
#define KOFF 288                // offset-conv K = 32*9
#define BPITCH (Kv + 8)         // 584: breaks LDS bank conflicts, keeps 16B align

typedef __attribute__((ext_vector_type(16))) __bf16 v16bf;
typedef __attribute__((ext_vector_type(8)))  __bf16 v8bf;
typedef __attribute__((ext_vector_type(8)))  float  v8f;

__device__ __forceinline__ float sigmoidf_(float x) { return 1.0f / (1.0f + __expf(-x)); }

__device__ __forceinline__ v8f wmma_bf16(v16bf a, v16bf b, v8f c) {
    return __builtin_amdgcn_wmma_f32_16x16x32_bf16(false, a, false, b, (short)0, c,
                                                   false, false);
}

// ---------------------------------------------------------------------------
// Kernel 0: prep bf16 weights.
//   wbf    [64][576]  : w_reg straight copy (k = (g*16+c)*9+kk == natural flat)
//   woffbf [32][288]  : w_off rows 0..17, zero-padded to M=32
//   wmodbf [48][576]  : w_mod rows 0..35, zero-padded to M=48
// ---------------------------------------------------------------------------
__global__ void prep_weights(const float* __restrict__ wreg,
                             const float* __restrict__ woff,
                             const float* __restrict__ wmod,
                             __bf16* __restrict__ wbf,
                             __bf16* __restrict__ woffbf,
                             __bf16* __restrict__ wmodbf) {
    int i = blockIdx.x * 256 + threadIdx.x;
    if (i < Ov * Kv) {
        wbf[i] = (__bf16)wreg[i];
    } else if (i < Ov * Kv + 32 * KOFF) {
        int j = i - Ov * Kv;
        int r = j / KOFF;
        woffbf[j] = (r < 18) ? (__bf16)woff[j] : (__bf16)0.0f;
    } else if (i < Ov * Kv + 32 * KOFF + 48 * Kv) {
        int j = i - (Ov * Kv + 32 * KOFF);
        int r = j / Kv;
        wmodbf[j] = (r < 36) ? (__bf16)wmod[j] : (__bf16)0.0f;
    }
}

// ---------------------------------------------------------------------------
// Kernel 1: offset conv + modulator conv as implicit GEMM on WMMA.
// Block = 256 threads (8 wave32), owns 16 consecutive pixels of one batch b
// (W % 16 == 0, tiles never cross rows).
//   Stage 1: bf16 im2col tiles in LDS, n-major:
//            smW[n][k], smS[n][k], k = ch*9 + tap, K = 576 each.
//   Stage 2: 11 wave-tile jobs over 8 waves:
//     jobs 0..2 : mask  = 2*sigmoid(Wmod[48x576] x smW + b_mod), rows m0=16*job
//     jobs 3..10: offs  = 80*sigmoid(Woff[32x288] x B_g + b_off) - 40,
//                 g = (job-3)/2, mt = (job-3)&1, B_g rows = smW[g*144..+144)
//                 ++ smS[g*144..+144) (144 is 8-aligned -> every 8-elem WMMA
//                 fragment run sits in exactly one tile).
// ---------------------------------------------------------------------------
__global__ __launch_bounds__(256) void conv_wmma(
    const float* __restrict__ warp, const float* __restrict__ src,
    const float* __restrict__ b_off, const float* __restrict__ b_mod,
    const __bf16* __restrict__ woffbf, const __bf16* __restrict__ wmodbf,
    float* __restrict__ offb, float* __restrict__ maskb) {
    __shared__ __bf16 smW[16][BPITCH];
    __shared__ __bf16 smS[16][BPITCH];

    int tid  = threadIdx.x;
    int blk  = blockIdx.x;
    int b    = blk / (HWv / 16);
    int tile = blk % (HWv / 16);
    int p0   = tile * 16;

    // ---- Stage 1: im2col fill (coalesced along n) ----
    for (int it = tid; it < Kv * 16; it += 256) {
        int n  = it & 15;
        int k  = it >> 4;            // 0..575
        int ch = k / KKv;
        int t  = k - ch * KKv;
        int p  = p0 + n;
        int h = p / Wv, w = p % Wv;
        int yy = h - 1 + t / 3, xx = w - 1 + t % 3;
        bool ok = (yy >= 0 && yy < Hv && xx >= 0 && xx < Wv);
        long idx = ((long)b * Cv + ch) * HWv + yy * Wv + xx;
        smW[n][k] = ok ? (__bf16)warp[idx] : (__bf16)0.0f;
        smS[n][k] = ok ? (__bf16)src[idx]  : (__bf16)0.0f;
    }
    __syncthreads();

    // ---- Stage 2: WMMA tile jobs ----
    int wave = tid >> 5;
    int lane = tid & 31;
    int lr   = lane & 15;   // M row (A) / N col (B,C)
    int hi   = lane >> 4;   // K-half select per 16-bit fragment layout

    for (int job = wave; job < 11; job += 8) {
        v8f acc = {};
        if (job < 3) {
            // -------- modulator conv tile: rows m0..m0+15 of padded 48 --------
            int m0 = job * 16;
            const __bf16* arow = wmodbf + (long)(m0 + lr) * Kv;
#pragma unroll 2
            for (int kc = 0; kc < Kv; kc += 32) {
                v8bf alo = *(const v8bf*)(arow + kc + hi * 8);
                v8bf ahi = *(const v8bf*)(arow + kc + 16 + hi * 8);
                v8bf blo = *(const v8bf*)(&smW[lr][kc + hi * 8]);
                v8bf bhi = *(const v8bf*)(&smW[lr][kc + 16 + hi * 8]);
                v16bf a, bm;
#pragma unroll
                for (int e = 0; e < 8; e++) {
                    a[e] = alo[e];  a[e + 8] = ahi[e];
                    bm[e] = blo[e]; bm[e + 8] = bhi[e];
                }
                acc = wmma_bf16(a, bm, acc);
            }
#pragma unroll
            for (int r = 0; r < 8; r++) {
                int oc = m0 + r + 8 * hi;           // global channel g*9 + kk
                if (oc < 36) {
                    float v = 2.0f * sigmoidf_(acc[r] + b_mod[oc]);
                    maskb[(long)(b * 36 + oc) * HWv + p0 + lr] = v;
                }
            }
        } else {
            // -------- offset conv tile for group g --------
            int j  = job - 3;
            int g  = j >> 1;
            int mt = j & 1;
            int kg = g * 144;                        // group row base in im2col
            const __bf16* arow = woffbf + (long)(mt * 16 + lr) * KOFF;
#pragma unroll 1
            for (int kc = 0; kc < KOFF; kc += 32) {
                int k0 = kc + hi * 8;
                int k1 = kc + 16 + hi * 8;
                const __bf16* b0 = (k0 < 144) ? &smW[lr][kg + k0] : &smS[lr][kg + k0 - 144];
                const __bf16* b1 = (k1 < 144) ? &smW[lr][kg + k1] : &smS[lr][kg + k1 - 144];
                v8bf alo = *(const v8bf*)(arow + k0);
                v8bf ahi = *(const v8bf*)(arow + k1);
                v8bf blo = *(const v8bf*)b0;
                v8bf bhi = *(const v8bf*)b1;
                v16bf a, bm;
#pragma unroll
                for (int e = 0; e < 8; e++) {
                    a[e] = alo[e];  a[e + 8] = ahi[e];
                    bm[e] = blo[e]; bm[e + 8] = bhi[e];
                }
                acc = wmma_bf16(a, bm, acc);
            }
#pragma unroll
            for (int r = 0; r < 8; r++) {
                int m = mt * 16 + r + 8 * hi;        // output channel 2*kk + d
                if (m < 18) {
                    float v = 80.0f * sigmoidf_(acc[r] + b_off[m]) - 40.0f;
                    int kk = m >> 1, d = m & 1;
                    offb[(((long)(b * Gv + g) * KKv + kk) * 2 + d) * HWv + p0 + lr] = v;
                }
            }
        }
    }
}

// ---------------------------------------------------------------------------
// Kernel 2: fused deformable sampling + final WMMA GEMM (unchanged; codegen
// already ideal: b128 A loads + ds_load_b128 B + 2 WMMA per unrolled iter).
// ---------------------------------------------------------------------------
__global__ __launch_bounds__(128) void fused_sample_gemm(
    const float* __restrict__ warp, const float* __restrict__ offb,
    const float* __restrict__ maskb, const __bf16* __restrict__ wbf,
    float* __restrict__ out) {
    __shared__ __bf16 smB[16][BPITCH];

    int tid  = threadIdx.x;
    int blk  = blockIdx.x;
    int b    = blk / (HWv / 16);
    int tile = blk % (HWv / 16);
    int p0   = tile * 16;

    // ---- Stage 1: deformable sampling into LDS ----
    for (int item = tid; item < Gv * KKv * 16; item += 128) {
        int n  = item & 15;
        int t  = item >> 4;        // 0..35
        int g  = t / KKv;
        int kk = t - g * KKv;
        int p  = p0 + n;
        int h = p / Wv, w = p % Wv;

        long obase = ((long)(b * Gv + g) * KKv + kk) * 2 * HWv + p;
        float dy = offb[obase];
        float dx = offb[obase + HWv];
        float m  = maskb[((long)(b * Gv + g) * KKv + kk) * HWv + p];

        float y = (float)(h - 1 + kk / 3) + dy;
        float x = (float)(w - 1 + kk % 3) + dx;
        float yf = floorf(y), xf = floorf(x);
        float ly = y - yf, lx = x - xf;
        int y0 = (int)yf, x0 = (int)xf;
        int y1 = y0 + 1, x1 = x0 + 1;
        float vy0 = (y0 >= 0 && y0 < Hv) ? 1.0f : 0.0f;
        float vy1 = (y1 >= 0 && y1 < Hv) ? 1.0f : 0.0f;
        float vx0 = (x0 >= 0 && x0 < Wv) ? 1.0f : 0.0f;
        float vx1 = (x1 >= 0 && x1 < Wv) ? 1.0f : 0.0f;
        int y0c = min(max(y0, 0), Hv - 1), y1c = min(max(y1, 0), Hv - 1);
        int x0c = min(max(x0, 0), Wv - 1), x1c = min(max(x1, 0), Wv - 1);
        float w00 = (1.0f - ly) * (1.0f - lx) * vy0 * vx0 * m;
        float w01 = (1.0f - ly) * lx          * vy0 * vx1 * m;
        float w10 = ly          * (1.0f - lx) * vy1 * vx0 * m;
        float w11 = ly          * lx          * vy1 * vx1 * m;
        int i00 = y0c * Wv + x0c, i01 = y0c * Wv + x1c;
        int i10 = y1c * Wv + x0c, i11 = y1c * Wv + x1c;

        const float* img = warp + (long)(b * Gv + g) * CGv * HWv;
        int kbase = g * 144 + kk;
#pragma unroll 4
        for (int c = 0; c < CGv; c++) {
            const float* ip = img + (long)c * HWv;
            float s = w00 * ip[i00] + w01 * ip[i01] + w10 * ip[i10] + w11 * ip[i11];
            smB[n][kbase + c * KKv] = (__bf16)s;
        }
    }
    __syncthreads();

    // ---- Stage 2: per-wave 16x16 tile, K = 576 in 18 chunks of 32 (bf16) ----
    int wave = tid >> 5;
    int lane = tid & 31;
    int lr   = lane & 15;
    int hi   = lane >> 4;

    v8f acc = {};
    const __bf16* arow = wbf + (long)(wave * 16 + lr) * Kv;
#pragma unroll 2
    for (int kc = 0; kc < Kv; kc += 32) {
        v8bf alo = *(const v8bf*)(arow + kc + hi * 8);
        v8bf ahi = *(const v8bf*)(arow + kc + 16 + hi * 8);
        v8bf blo = *(const v8bf*)(&smB[lr][kc + hi * 8]);
        v8bf bhi = *(const v8bf*)(&smB[lr][kc + 16 + hi * 8]);
        v16bf a, bm;
#pragma unroll
        for (int e = 0; e < 8; e++) {
            a[e] = alo[e];  a[e + 8] = ahi[e];
            bm[e] = blo[e]; bm[e + 8] = bhi[e];
        }
        acc = wmma_bf16(a, bm, acc);
    }

#pragma unroll
    for (int r = 0; r < 8; r++) {
        int oc = wave * 16 + r + hi * 8;
        out[(long)(b * Ov + oc) * HWv + p0 + lr] = acc[r];
    }
}

// ---------------------------------------------------------------------------
// Launch. Inputs (setup_inputs order, all fp32):
// 0:warp_ref 1:source 2:w_off 3:b_off 4:w_mod 5:b_mod 6:w_reg
// ws: offb [4*4*9*2*25600 f32] | maskb [4*4*9*25600 f32] |
//     wbf [64*576 bf16] | woffbf [32*288 bf16] | wmodbf [48*576 bf16]
// total ~42.3 MB scratch (L2-resident on the 192 MB L2).
// ---------------------------------------------------------------------------
extern "C" void kernel_launch(void* const* d_in, const int* in_sizes, int n_in,
                              void* d_out, int out_size, void* d_ws, size_t ws_size,
                              hipStream_t stream) {
    const float* warp  = (const float*)d_in[0];
    const float* src   = (const float*)d_in[1];
    const float* w_off = (const float*)d_in[2];
    const float* b_off = (const float*)d_in[3];
    const float* w_mod = (const float*)d_in[4];
    const float* b_mod = (const float*)d_in[5];
    const float* w_reg = (const float*)d_in[6];
    float* out = (float*)d_out;

    float*  offb   = (float*)d_ws;
    float*  maskb  = offb + (size_t)Bv * Gv * KKv * 2 * HWv;
    __bf16* wbf    = (__bf16*)(maskb + (size_t)Bv * Gv * KKv * HWv);
    __bf16* woffbf = wbf + (size_t)Ov * Kv;
    __bf16* wmodbf = woffbf + (size_t)32 * KOFF;

    int prep_n = Ov * Kv + 32 * KOFF + 48 * Kv;   // 73728
    prep_weights<<<(prep_n + 255) / 256, 256, 0, stream>>>(w_reg, w_off, w_mod,
                                                           wbf, woffbf, wmodbf);
    conv_wmma<<<Bv * (HWv / 16), 256, 0, stream>>>(warp, src, b_off, b_mod,
                                                   woffbf, wmodbf, offb, maskb);
    fused_sample_gemm<<<Bv * (HWv / 16), 128, 0, stream>>>(warp, offb, maskb, wbf, out);
}
